// NeRF_70471823393213
// MI455X (gfx1250) — matrix operand order repaired and checked by reading, not compiled, not run on hardware
//
#include <hip/hip_runtime.h>
#include <hip/hip_bf16.h>
#include <math.h>

// ---------------------------------------------------------------------------
// Fused NeRF-style per-point MLP chain for MI455X (gfx1250, wave32, WMMA).
// One wave processes 16 points; every layer with K>=24 runs as
// v_wmma_f32_16x16x32_bf16 (bf16 in, f32 accum). Weights live in LDS as
// zero-padded bf16 A-tiles, activations staged per-wave in [point][K] bf16.
// v2: view-invariant w1 K-steps hoisted (64->40 WMMAs), both lane halves
// used for all scalar glue, A-fragment/dot-weight hoisting, no copy phases.
// ---------------------------------------------------------------------------

typedef __attribute__((ext_vector_type(16))) __bf16 v16bf;
typedef __attribute__((ext_vector_type(8)))  float  v8f;

#define NVIEW 4
#define VOXC  8
#define DIN   35

#define WAVES 4
#define PTS_PER_BLOCK (WAVES * 16)

// ---- per-wave LDS scratch (bytes) -----------------------------------------
#define PW_BYTES 12288
#define OFF_CAT 0      // cat[v][16][64] bf16 = 8192 B  (phases 1-2)
#define OFF_BFC 0      // gsum B [16][32] bf16 = 1024 B (phase 3-4, reuse)
#define OFF_WF  0      // wf [16][128] bf16 = 4096 B    (phase 4-7, reuse)
#define OFF_X   4096   // x  [16][64]  bf16 = 2048 B    (phase 5+)
#define OFF_G   8192   // g[v][16][32] bf16 = 4096 B    (phase 2-3)
#define OFF_VI  8192   // voximg [16][32] bf16 = 1024 B (phase 4+, reuse)

// ---- block-shared weight region -------------------------------------------
#define WOFF    (WAVES * PW_BYTES)          // 49152
#define WO_GLOB (WOFF + 0)                  // 32x64  bf16 = 4096
#define WO_LR0  (WOFF + 4096)               // 64x32  bf16 = 4096
#define WO_W1   (WOFF + 8192)               // 64x128 bf16 = 16384
#define WO_FC   (WOFF + 24576)              // 16x32  bf16 = 1024
#define WO_FEAT (WOFF + 25600)              // 16x64  bf16 = 2048
#define WO_F32  (WOFF + 27648)              // small f32 vectors
// f32 area element indices
#define F_WVIEW 0      // 19x4 = 76
#define F_BVIEW 76     // 19
#define F_BGLOB 95     // 32
#define F_WAGG  127    // 32
#define F_BAGG  159    // 1
#define F_BFC   160    // 16
#define F_BLR0  176    // 64
#define F_WSIG  240    // 64
#define F_BSIG  304    // 1
#define F_BW1   305    // 64
#define F_WW2   369    // 64
#define F_BW2   433    // 1
#define F_BFEAT 434    // 8  -> 442 total
#define SMEM_BYTES (WO_F32 - 0 + 442 * 4)   // 78568

// ---------------------------------------------------------------------------
__device__ __forceinline__ v8f vzero8() {
  v8f z = {0.f, 0.f, 0.f, 0.f, 0.f, 0.f, 0.f, 0.f};
  return z;
}

// A fragment (16 x 32 bf16, row-major LDS weight with leading dim ldk).
// Lane layout: m = m0 + (lane&15); j<8 -> k = k0 + 8*hi + j;
//              j>=8 -> k = k0 + 8*hi + 16 + (j-8).
__device__ __forceinline__ v16bf ldsA(const __bf16* W, int ldk, int m0, int k0, int lane) {
  const int m  = m0 + (lane & 15);
  const int kb = k0 + ((lane >> 4) << 3);
  const __bf16* p0 = W + m * ldk + kb;
  v16bf a;
#pragma unroll
  for (int j = 0; j < 8; ++j) { a[j] = p0[j]; a[j + 8] = p0[16 + j]; }
  return a;
}

// B fragment (32 x 16 bf16) from activations staged as [point][Kpad].
// Lane layout: n = lane&15; j<8 -> k = k0 + 8*hi + j; j>=8 -> k = k0 + 8*hi + 16 + (j-8).
__device__ __forceinline__ v16bf ldsB(const __bf16* act, int ldk, int k0, int lane) {
  const int nn = lane & 15;
  const int kb = k0 + ((lane >> 4) << 3);
  const __bf16* p = act + nn * ldk + kb;
  v16bf b;
#pragma unroll
  for (int j = 0; j < 8; ++j) { b[j] = p[j]; b[j + 8] = p[16 + j]; }
  return b;
}

__device__ __forceinline__ v8f wmma_bf16(v16bf a, v16bf b, v8f c) {
  return __builtin_amdgcn_wmma_f32_16x16x32_bf16(false, a, false, b, (short)0, c, false, false);
}

// Hoist this lane's slice of a 1-row dot-weight vector into registers:
// lane owns m = t*16 + r + 8*hi.
template <int NT>
__device__ __forceinline__ void load_dotw(float* wreg, const float* wv, int hi) {
#pragma unroll
  for (int t = 0; t < NT; ++t)
#pragma unroll
    for (int r = 0; r < 8; ++r) wreg[t * 8 + r] = wv[t * 16 + r + 8 * hi];
}

// Dot over the M dimension of WMMA D accumulators; partner halves combined
// with a 16-xor shuffle (wave32). Result valid in all lanes, point n=lane&15.
template <int NT>
__device__ __forceinline__ float dot_rows_reg(const v8f (&acc)[NT], const float* wreg) {
  float s = 0.f;
#pragma unroll
  for (int t = 0; t < NT; ++t)
#pragma unroll
    for (int r = 0; r < 8; ++r) s += acc[t][r] * wreg[t * 8 + r];
  s += __shfl_xor(s, 16, 32);
  return s;
}

__device__ __forceinline__ void stage_w_bf16(__bf16* dst, const float* src, int M,
                                             int Ksrc, int Kpad, int Mvalid,
                                             int tid, int nth) {
  const int tot = M * Kpad;
  for (int i = tid; i < tot; i += nth) {
    const int m = i / Kpad, k = i - m * Kpad;
    const float v = (m < Mvalid && k < Ksrc) ? src[m * Ksrc + k] : 0.f;
    dst[i] = (__bf16)v;
  }
}

__device__ __forceinline__ void stage_f32(float* dst, const float* src, int n,
                                          int tid, int nth) {
  for (int i = tid; i < n; i += nth) dst[i] = src[i];
}

// ---------------------------------------------------------------------------
__global__ void __launch_bounds__(128)
nerf_fused_wmma(const float* __restrict__ vox, const float* __restrict__ rgbs,
                const float* __restrict__ Wview, const float* __restrict__ bview,
                const float* __restrict__ Wglob, const float* __restrict__ bglob,
                const float* __restrict__ Wagg,  const float* __restrict__ bagg,
                const float* __restrict__ Wfc,   const float* __restrict__ bfc_,
                const float* __restrict__ Wlr0,  const float* __restrict__ blr0,
                const float* __restrict__ Wsig,  const float* __restrict__ bsig,
                const float* __restrict__ Ww1,   const float* __restrict__ bw1,
                const float* __restrict__ Ww2,   const float* __restrict__ bw2,
                const float* __restrict__ Wfeat, const float* __restrict__ bfeat,
                float* __restrict__ out, int N) {
  extern __shared__ char smem[];
  const int tid  = threadIdx.x;
  const int wave = tid >> 5;
  const int lane = tid & 31;
  const int n    = lane & 15;
  const int hi   = (lane >> 4) & 1;
  const int p    = blockIdx.x * PTS_PER_BLOCK + wave * 16 + n;

  // ---- phase 0: cooperative weight staging (padded bf16 A-tiles + f32) ----
  stage_w_bf16((__bf16*)(smem + WO_GLOB), Wglob, 32, 57, 64, 32, tid, 128);
  stage_w_bf16((__bf16*)(smem + WO_LR0),  Wlr0,  64, 24, 32, 64, tid, 128);
  stage_w_bf16((__bf16*)(smem + WO_W1),   Ww1,   64, 111, 128, 64, tid, 128);
  stage_w_bf16((__bf16*)(smem + WO_FC),   Wfc,   16, 32, 32, 16, tid, 128);
  stage_w_bf16((__bf16*)(smem + WO_FEAT), Wfeat, 16, 64, 64, 8, tid, 128);
  float* f32w = (float*)(smem + WO_F32);
  stage_f32(f32w + F_WVIEW, Wview, 76, tid, 128);
  stage_f32(f32w + F_BVIEW, bview, 19, tid, 128);
  stage_f32(f32w + F_BGLOB, bglob, 32, tid, 128);
  stage_f32(f32w + F_WAGG,  Wagg,  32, tid, 128);
  stage_f32(f32w + F_BAGG,  bagg,   1, tid, 128);
  stage_f32(f32w + F_BFC,   bfc_,  16, tid, 128);
  stage_f32(f32w + F_BLR0,  blr0,  64, tid, 128);
  stage_f32(f32w + F_WSIG,  Wsig,  64, tid, 128);
  stage_f32(f32w + F_BSIG,  bsig,   1, tid, 128);
  stage_f32(f32w + F_BW1,   bw1,   64, tid, 128);
  stage_f32(f32w + F_WW2,   Ww2,   64, tid, 128);
  stage_f32(f32w + F_BW2,   bw2,    1, tid, 128);
  stage_f32(f32w + F_BFEAT, bfeat,  8, tid, 128);
  __syncthreads();

  char* ms = smem + wave * PW_BYTES;
  __bf16* cat0 = (__bf16*)(ms + OFF_CAT);

  // ---- phase 1: view MLP, img, var/avg -> cat[v][n][64] -------------------
  // Both halves own point n; lo half does channels 0..9, hi half 10..18.
  {
    const int cb = hi * 10;
    float s1[10], s2[10];
#pragma unroll
    for (int i = 0; i < 10; ++i) { s1[i] = 0.f; s2[i] = 0.f; }
#pragma unroll
    for (int v = 0; v < NVIEW; ++v) {
      const float* row = rgbs + ((size_t)v * N + p) * DIN;
      const float d0 = row[31], d1 = row[32], d2 = row[33], d3 = row[34];
      __bf16* catv = cat0 + v * 1024 + n * 64;
#pragma unroll
      for (int i = 0; i < 10; ++i) {
        const int c = cb + i;
        if (c < 19) {
          const float* wr = f32w + F_WVIEW + c * 4;
          float h = fmaf(wr[0], d0, fmaf(wr[1], d1, fmaf(wr[2], d2,
                    fmaf(wr[3], d3, f32w[F_BVIEW + c]))));
          h = fmaxf(h, 0.f);
          const float im = row[12 + c] + h;
          catv[c] = (__bf16)im;
          s1[i] += im; s2[i] += im * im;
        }
      }
    }
#pragma unroll
    for (int i = 0; i < 10; ++i) {
      const int c = cb + i;
      if (c < 19) {
        const float av = 0.25f * s1[i];
        const float vr = (s2[i] - 4.f * av * av) * (1.f / 3.f);  // ddof=1
#pragma unroll
        for (int v = 0; v < NVIEW; ++v) {
          __bf16* catv = cat0 + v * 1024 + n * 64;
          catv[19 + c] = (__bf16)vr;
          catv[38 + c] = (__bf16)av;
        }
      }
    }
#pragma unroll
    for (int v = 0; v < NVIEW; ++v) {
      __bf16* catv = cat0 + v * 1024 + n * 64;
#pragma unroll
      for (int i = 0; i < 4; ++i) {
        const int c = 57 + hi * 4 + i;               // lo: 57..60, hi: 61..63
        if (c < 64) catv[c] = (__bf16)0.f;
      }
    }
  }

  // ---- phase 2: glob GEMM (32x64 @ 64x16) per view, aggw logits -----------
  // A fragments are view-invariant: hoist across the view loop.
  __bf16* gl = (__bf16*)(ms + OFF_G);
  float law[NVIEW];
  {
    float agg_r[16];
    load_dotw<2>(agg_r, f32w + F_WAGG, hi);
    v16bf gA[4];
#pragma unroll
    for (int mt = 0; mt < 2; ++mt)
#pragma unroll
      for (int ks = 0; ks < 2; ++ks)
        gA[mt * 2 + ks] = ldsA((const __bf16*)(smem + WO_GLOB), 64, mt * 16, ks * 32, lane);
#pragma unroll
    for (int v = 0; v < NVIEW; ++v) {
      v8f acc[2] = {vzero8(), vzero8()};
#pragma unroll
      for (int ks = 0; ks < 2; ++ks) {
        v16bf B = ldsB(cat0 + v * 1024, 64, ks * 32, lane);
#pragma unroll
        for (int mt = 0; mt < 2; ++mt)
          acc[mt] = wmma_bf16(gA[mt * 2 + ks], B, acc[mt]);
      }
#pragma unroll
      for (int mt = 0; mt < 2; ++mt) {
#pragma unroll
        for (int r = 0; r < 8; ++r) {
          const int m = mt * 16 + r + 8 * hi;
          const float gv = fmaxf(acc[mt][r] + f32w[F_BGLOB + m], 0.f);
          acc[mt][r] = gv;
          gl[v * 512 + n * 32 + m] = (__bf16)gv;
        }
      }
      law[v] = fmaxf(dot_rows_reg<2>(acc, agg_r) + f32w[F_BAGG], 0.f);
    }
  }

  // ---- phase 3: softmax over views, gsum -> B tile for fc -----------------
  float aw[NVIEW];
  {
    float mx = fmaxf(fmaxf(law[0], law[1]), fmaxf(law[2], law[3]));
    float se = 0.f;
#pragma unroll
    for (int v = 0; v < NVIEW; ++v) { aw[v] = expf(law[v] - mx); se += aw[v]; }
    const float inv = 1.f / se;
#pragma unroll
    for (int v = 0; v < NVIEW; ++v) aw[v] *= inv;
  }
  __bf16* bfcB = (__bf16*)(ms + OFF_BFC);
  {
    const int gc0 = hi * 16;                         // lo: 0..15, hi: 16..31
#pragma unroll
    for (int ci = 0; ci < 16; ++ci) {
      const int c = gc0 + ci;
      float gs = 0.f;
#pragma unroll
      for (int v = 0; v < NVIEW; ++v) gs += (float)gl[v * 512 + n * 32 + c] * aw[v];
      bfcB[n * 32 + c] = (__bf16)gs;
    }
  }

  // ---- phase 4: fc GEMM -> im_feat; build vox_img B tile AND wf slice -----
  __bf16* vi = (__bf16*)(ms + OFF_VI);
  __bf16* wf = (__bf16*)(ms + OFF_WF);
  {
    v8f fca = vzero8();
    v16bf A = ldsA((const __bf16*)(smem + WO_FC), 32, 0, 0, lane);
    v16bf B = ldsB(bfcB, 32, 0, lane);
    fca = wmma_bf16(A, B, fca);
#pragma unroll
    for (int r = 0; r < 8; ++r) {
      const int m = r + 8 * hi;
      const __bf16 imf = (__bf16)fmaxf(fca[r] + f32w[F_BFC + m], 0.f);
      vi[n * 32 + 8 + m] = imf;                      // im_feat -> vox_img ch 8..23
      wf[n * 128 + 64 + 8 + m] = imf;                // and wf ch 72..87
    }
    if (!hi) {
#pragma unroll
      for (int c = 0; c < 8; ++c) {
        const __bf16 vv = (__bf16)vox[(size_t)p * VOXC + c];
        vi[n * 32 + c] = vv;
        wf[n * 128 + 64 + c] = vv;                   // wf ch 64..71
      }
    }
#pragma unroll
    for (int ci = 0; ci < 4; ++ci)                   // vox_img pad 24..31
      vi[n * 32 + 24 + hi * 4 + ci] = (__bf16)0.f;
#pragma unroll
    for (int ci = 0; ci < 9; ++ci) {                 // wf pad 111..127
      const int c = 111 + hi * 9 + ci;
      if (c < 128) wf[n * 128 + c] = (__bf16)0.f;
    }
  }

  // ---- phase 5: lr0 GEMM (64x32 @ 32x16) -> x (dual-stored), sigma --------
  __bf16* xl = (__bf16*)(ms + OFF_X);
  v8f xa[4];
  {
    v16bf B = ldsB(vi, 32, 0, lane);
#pragma unroll
    for (int mt = 0; mt < 4; ++mt) {
      v16bf A = ldsA((const __bf16*)(smem + WO_LR0), 32, mt * 16, 0, lane);
      xa[mt] = wmma_bf16(A, B, vzero8());
    }
  }
#pragma unroll
  for (int mt = 0; mt < 4; ++mt) {
#pragma unroll
    for (int r = 0; r < 8; ++r) {
      const int m = mt * 16 + r + 8 * hi;
      const float xv = fmaxf(xa[mt][r] + f32w[F_BLR0 + m], 0.f);
      xa[mt][r] = xv;
      const __bf16 xb = (__bf16)xv;
      xl[n * 64 + m] = xb;                           // feat-GEMM B tile
      wf[n * 128 + m] = xb;                          // wf ch 0..63
    }
  }
  {
    float sig_r[32];
    load_dotw<4>(sig_r, f32w + F_WSIG, hi);
    float sg = dot_rows_reg<4>(xa, sig_r) + f32w[F_BSIG];
    sg = (sg > 20.f) ? sg : log1pf(expf(sg));        // softplus
    if (!hi) out[p] = sg;
  }

  // ---- phase 7: w1 GEMM (64x128 @ 128x16): shared K 0..63, per-view rest --
  float lw[NVIEW];
  {
    float w2_r[32];
    load_dotw<4>(w2_r, f32w + F_WW2, hi);
    v8f hpart[4];                                    // view-invariant partial
#pragma unroll
    for (int ks = 0; ks < 2; ++ks) {
      v16bf B = ldsB(wf, 128, ks * 32, lane);
#pragma unroll
      for (int mt = 0; mt < 4; ++mt) {
        v16bf A = ldsA((const __bf16*)(smem + WO_W1), 128, mt * 16, ks * 32, lane);
        hpart[mt] = (ks == 0) ? wmma_bf16(A, B, vzero8())
                              : wmma_bf16(A, B, hpart[mt]);
      }
    }
#pragma unroll
    for (int v = 0; v < NVIEW; ++v) {
      {                                              // stage frd slice, split
        const float* row = rgbs + ((size_t)v * N + p) * DIN;
#pragma unroll
        for (int ci = 0; ci < 12; ++ci) {
          const int c = hi * 12 + ci;
          if (c < 23) wf[n * 128 + 88 + c] = (__bf16)row[12 + c];
        }
      }
      v8f ha[4];
#pragma unroll
      for (int mt = 0; mt < 4; ++mt) {
        v16bf A2 = ldsA((const __bf16*)(smem + WO_W1), 128, mt * 16, 64, lane);
        v16bf B2 = ldsB(wf, 128, 64, lane);
        v8f hv = wmma_bf16(A2, B2, hpart[mt]);       // C!=D: no reg copies
        v16bf A3 = ldsA((const __bf16*)(smem + WO_W1), 128, mt * 16, 96, lane);
        v16bf B3 = ldsB(wf, 128, 96, lane);
        hv = wmma_bf16(A3, B3, hv);
#pragma unroll
        for (int r = 0; r < 8; ++r) {
          const int m = mt * 16 + r + 8 * hi;
          hv[r] = fmaxf(hv[r] + f32w[F_BW1 + m], 0.f);
        }
        ha[mt] = hv;
      }
      lw[v] = fmaxf(dot_rows_reg<4>(ha, w2_r) + f32w[F_BW2], 0.f);
    }
  }

  // ---- phase 8: softmax wt, weighted rgb sum (split), feat GEMM, store ----
  float wt[NVIEW];
  {
    float mx = fmaxf(fmaxf(lw[0], lw[1]), fmaxf(lw[2], lw[3]));
    float se = 0.f;
#pragma unroll
    for (int v = 0; v < NVIEW; ++v) { wt[v] = expf(lw[v] - mx); se += wt[v]; }
    const float inv = 1.f / se;
#pragma unroll
    for (int v = 0; v < NVIEW; ++v) wt[v] *= inv;
  }
  v8f fa = vzero8();
#pragma unroll
  for (int ks = 0; ks < 2; ++ks) {
    v16bf A = ldsA((const __bf16*)(smem + WO_FEAT), 64, 0, ks * 32, lane);
    v16bf B = ldsB(xl, 64, ks * 32, lane);
    fa = wmma_bf16(A, B, fa);
  }
  {
    float* of = out + N + (size_t)p * 39;
#pragma unroll
    for (int ci = 0; ci < 16; ++ci) {                // lo: ch 0..15, hi: 16..30
      const int c = hi * 16 + ci;
      if (c < 31) {
        float rs = 0.f;
#pragma unroll
        for (int v = 0; v < NVIEW; ++v)
          rs += rgbs[((size_t)v * N + p) * DIN + c] * wt[v];
        of[c] = rs;
      }
    }
    if (!hi) {                                       // hi==0 lanes hold m=r=0..7
#pragma unroll
      for (int r = 0; r < 8; ++r)
        of[31 + r] = fmaxf(fa[r] + f32w[F_BFEAT + r], 0.f);
    }
  }
}

// ---------------------------------------------------------------------------
extern "C" void kernel_launch(void* const* d_in, const int* in_sizes, int n_in,
                              void* d_out, int out_size, void* d_ws, size_t ws_size,
                              hipStream_t stream) {
  const float* vox   = (const float*)d_in[0];
  const float* rgbs  = (const float*)d_in[1];
  const float* Wview = (const float*)d_in[2];
  const float* bview = (const float*)d_in[3];
  const float* Wglob = (const float*)d_in[4];
  const float* bglob = (const float*)d_in[5];
  const float* Wagg  = (const float*)d_in[6];
  const float* bagg  = (const float*)d_in[7];
  const float* Wfc   = (const float*)d_in[8];
  const float* bfc   = (const float*)d_in[9];
  const float* Wlr0  = (const float*)d_in[10];
  const float* blr0  = (const float*)d_in[11];
  const float* Wsig  = (const float*)d_in[12];
  const float* bsig  = (const float*)d_in[13];
  const float* Ww1   = (const float*)d_in[14];
  const float* bw1   = (const float*)d_in[15];
  const float* Ww2   = (const float*)d_in[16];
  const float* bw2   = (const float*)d_in[17];
  const float* Wfeat = (const float*)d_in[18];
  const float* bfeat = (const float*)d_in[19];
  float* out = (float*)d_out;

  const int N = in_sizes[0] / VOXC;                       // 262144
  const int blocks = (N + PTS_PER_BLOCK - 1) / PTS_PER_BLOCK;

  hipFuncSetAttribute(reinterpret_cast<const void*>(nerf_fused_wmma),
                      hipFuncAttributeMaxDynamicSharedMemorySize, SMEM_BYTES);

  nerf_fused_wmma<<<blocks, 128, SMEM_BYTES, stream>>>(
      vox, rgbs, Wview, bview, Wglob, bglob, Wagg, bagg, Wfc, bfc,
      Wlr0, blr0, Wsig, bsig, Ww1, bw1, Ww2, bw2, Wfeat, bfeat, out, N);
}
